// MultiScaleAttn_87505663689325
// MI455X (gfx1250) — matrix operand order repaired and checked by reading, not compiled, not run on hardware
//
#include <hip/hip_runtime.h>
#include <hip/hip_bf16.h>

// MI455X / gfx1250 implementation (wave32, WMMA, async-to-LDS staging).
// Pipeline: pack_weights -> conv1x1_wmma -> l2norm_inplace ->
//           corr_flow_wmma (WMMA correlation + argmax + subpixel) -> 3x warp_fold.
// Roofline: ~60MB unique traffic @ 23.3 TB/s => few microseconds; conv GEMM
// (1.2 GFLOP) and the 17x17 correlation both ride v_wmma_f32_16x16x32_f16.
// conv input tiles are DMA'd with GLOBAL_LOAD_ASYNC_TO_LDS_B32 (ASYNCcnt) and
// consumed after s_wait_asynccnt; LDS reuse is fenced with s_wait_dscnt.
// All bounds handling is branchless (clamp + mask).

typedef __attribute__((ext_vector_type(16))) _Float16 v16h;
typedef __attribute__((ext_vector_type(8)))  _Float16 v8h;
typedef __attribute__((ext_vector_type(8)))  float    v8f;

#define HW   64
#define CIN  128
#define COUT 32
#define PIX  (HW * HW)
#define PLANE ((size_t)HW * HW)

__device__ __forceinline__ int clampi(int v, int lo, int hi_) {
    return v < lo ? lo : (v > hi_ ? hi_ : v);
}

// wave-relative LDS byte offset of a __shared__ object (flat-LDS aperture:
// low 32 bits of the address, ISA 10.2)
__device__ __forceinline__ unsigned lds_off(const void* p) {
    return (unsigned)(unsigned long long)(uintptr_t)p;
}

// ---------------------------------------------------------------------------
// Pack W_in (32x1152, K = c*9 + ij) into per-(mtile, ij, kchunk) WMMA A
// fragments (f16). A 16x32 layout: lane L: M=L%16, hi=L/16; element e:
// e<8 -> K = hi*8 + e ; e>=8 -> K = 16 + hi*8 + (e-8). Lane's 16 halves are
// stored contiguously so the GEMM kernel loads one v16h per fragment.
// ---------------------------------------------------------------------------
__global__ void pack_weights(const float* __restrict__ W_in,
                             _Float16* __restrict__ aPack) {
    int blk  = blockIdx.x;            // (mt*9 + ij)*4 + kc, 72 blocks
    int kc   = blk & 3;
    int mtij = blk >> 2;
    int ij   = mtij % 9;
    int mt   = mtij / 9;
    int L    = threadIdx.x;
    int hi   = L >> 4;
    int m    = mt * 16 + (L & 15);
    _Float16* dst = aPack + (size_t)blk * 512 + (size_t)L * 16;
#pragma unroll
    for (int e = 0; e < 16; ++e) {
        int kl = (e < 8) ? (hi * 8 + e) : (16 + hi * 8 + (e - 8));
        int c  = kc * 32 + kl;
        dst[e] = (_Float16)W_in[(size_t)m * 1152 + (size_t)c * 9 + ij];
    }
}

// ---------------------------------------------------------------------------
// conv1x1 over 3x3-unfold columns as WMMA GEMM. One wave per 16-pixel strip.
// Each ij shift: DMA the 128ch x 16px f32 tile into LDS with
// global_load_async_to_lds_b32 (per-lane, clamped addresses), wait ASYNCcnt,
// then build B fragments (channels kc*32+hi*8+e for e<8 and kc*32+16+hi*8+(e-8) for e>=8: two 8-float runs in
// LDS) with the per-pixel zero-pad mask applied at f16 conversion time.
// ---------------------------------------------------------------------------
__global__ __launch_bounds__(32)
void conv1x1_wmma(const float* __restrict__ KVt4x, const float* __restrict__ KV04x,
                  const _Float16* __restrict__ aPack,
                  float* __restrict__ a_raw, float* __restrict__ b_raw) {
    __shared__ float sF[16 * CIN];      // [px][c]  8 KB staging tile

    int id    = blockIdx.x;             // 4 images x 256 strips
    int img   = id >> 8;
    int strip = id & 255;
    int y     = strip >> 2;
    int x0    = (strip & 3) << 4;

    const float* src = (img < 2) ? (KVt4x + (size_t)img * CIN * PLANE)
                                 : (KV04x + (size_t)(img - 2) * CIN * PLANE);
    float* dst = (img < 2) ? (a_raw + (size_t)img * COUT * PLANE)
                           : (b_raw + (size_t)(img - 2) * COUT * PLANE);

    int L  = threadIdx.x;
    int hi = L >> 4;
    int px = L & 15;

    unsigned ldsBase = lds_off(&sF[0]);

    v8f acc0 = {};
    v8f acc1 = {};

    for (int ij = 0; ij < 9; ++ij) {
        int dy = ij / 3 - 1, dx = ij % 3 - 1;
        int yy = y + dy;
        int xx = x0 + px + dx;
        float mk = ((yy >= 0) && (yy < HW) && (xx >= 0) && (xx < HW)) ? 1.0f : 0.0f;
        int yyc = clampi(yy, 0, HW - 1);
        int xxc = clampi(xx, 0, HW - 1);
        const float* sp = src + (size_t)yyc * HW + xxc;

        // previous iteration's ds reads must retire before we overwrite LDS
        // (DScnt and ASYNCcnt are mutually unordered)
        asm volatile("s_wait_dscnt 0x0" ::: "memory");

        // DMA stage: lane covers channels c = 2t + hi for its pixel column
#pragma unroll
        for (int t = 0; t < 64; ++t) {
            int c = t * 2 + hi;
            unsigned lo = ldsBase + (unsigned)(px * CIN + c) * 4u;
            const float* g = sp + (size_t)c * PLANE;
            asm volatile("global_load_async_to_lds_b32 %0, %1, off"
                         :: "v"(lo), "v"(g) : "memory");
        }
        asm volatile("s_wait_asynccnt 0x0" ::: "memory");

#pragma unroll
        for (int kc = 0; kc < 4; ++kc) {
            const float* fp = &sF[px * CIN + kc * 32 + hi * 8];
            v16h bfr;
#pragma unroll
            for (int e = 0; e < 16; ++e)
                bfr[e] = (_Float16)(fp[(e < 8) ? e : (e + 8)] * mk);
            v16h a0 = *(const v16h*)(aPack + (size_t)(ij * 4 + kc) * 512 + (size_t)L * 16);
            v16h a1 = *(const v16h*)(aPack + (size_t)((9 + ij) * 4 + kc) * 512 + (size_t)L * 16);
            acc0 = __builtin_amdgcn_wmma_f32_16x16x32_f16(
                false, a0, false, bfr, (short)0, acc0, false, false);
            acc1 = __builtin_amdgcn_wmma_f32_16x16x32_f16(
                false, a1, false, bfr, (short)0, acc1, false, false);
        }
    }
    // C/D layout: N = L%16, M = hi*8 + r
#pragma unroll
    for (int r = 0; r < 8; ++r) {
        int m = hi * 8 + r;
        dst[((size_t)m * HW + y) * HW + x0 + px]        = acc0[r];
        dst[((size_t)(16 + m) * HW + y) * HW + x0 + px] = acc1[r];
    }
}

// ---------------------------------------------------------------------------
// In-place L2 normalization over the 32 channels (per pixel).
// ---------------------------------------------------------------------------
__global__ void l2norm_inplace(float* __restrict__ a, float* __restrict__ b) {
    int id = blockIdx.x * 256 + threadIdx.x;     // 16384 threads
    float* base = (id < 8192) ? a : b;
    int r = id & 8191;
    float* p = base + (size_t)(r >> 12) * (COUT * PLANE) + (size_t)(r & 4095);
    float s = 0.f;
#pragma unroll
    for (int c = 0; c < COUT; ++c) { float v = p[(size_t)c * PLANE]; s += v * v; }
    float inv = 1.0f / fmaxf(sqrtf(s), 1e-12f);
#pragma unroll
    for (int c = 0; c < COUT; ++c) p[(size_t)c * PLANE] *= inv;
}

// ---------------------------------------------------------------------------
// WMMA correlation: one wave per (n, 16-pixel strip).
// A = a-tile (16 px x 32 ch). For each dv row, 3 B tiles of 16 b-pixels
// covering xb in [x0-16, x0+32) give every du in [-8,8] on the band:
// D[m][col], col = m + du + 16. D tiles spill to LDS; lanes scan their pixel's
// du-subset with first-max semantics; halves combine with lexicographic
// tie-break; 3x3 quadratic refit recomputes 9 masked dots.
// ---------------------------------------------------------------------------
__global__ __launch_bounds__(32)
void corr_flow_wmma(const float* __restrict__ a_norm, const float* __restrict__ b_norm,
                    const float* __restrict__ subAttn, float* __restrict__ maxValue,
                    float* __restrict__ flow) {
    __shared__ _Float16 sA[16 * 32];   // [px][c]
    __shared__ float    sC[16 * 48];   // [m][col]
    __shared__ float    sBest[32];
    __shared__ int      sD[32];

    int id    = blockIdx.x;            // 2 x 256
    int n     = id >> 8;
    int strip = id & 255;
    int y     = strip >> 2;
    int x0    = (strip & 3) << 4;
    int L     = threadIdx.x;
    int hi    = L >> 4;
    int px    = L & 15;

    const float* ap = a_norm + (size_t)n * COUT * PLANE;
    const float* bp = b_norm + (size_t)n * COUT * PLANE;

    // stage a-tile (coalesced), convert to f16
#pragma unroll
    for (int t = 0; t < 16; ++t) {
        int c = t * 2 + hi;
        sA[px * 32 + c] = (_Float16)ap[(size_t)c * PLANE + (size_t)y * HW + x0 + px];
    }
    __syncthreads();

    // build A fragment: lane's halves = channels [hi*8..+7] and [16+hi*8..+7]
    v16h afr;
    {
        const _Float16* base = sA + px * 32;
        v8h lo  = *(const v8h*)(base + hi * 8);
        v8h hi8 = *(const v8h*)(base + 16 + hi * 8);
#pragma unroll
        for (int e = 0; e < 8; ++e) { afr[e] = lo[e]; afr[8 + e] = hi8[e]; }
    }

    float best = -3.0e38f;
    int   bestD = 0;

    for (int dv = -8; dv <= 8; ++dv) {
        int yy = y + dv;
        float okyf = ((yy >= 0) && (yy < HW)) ? 1.0f : 0.0f;
        int yyc = clampi(yy, 0, HW - 1);
#pragma unroll
        for (int t = 0; t < 3; ++t) {
            int xb = x0 + (t - 1) * 16 + px;
            float okf = okyf * (((xb >= 0) && (xb < HW)) ? 1.0f : 0.0f);
            int xbc = clampi(xb, 0, HW - 1);
            const float* q = bp + (size_t)yyc * HW + xbc;
            v16h bfr;
#pragma unroll
            for (int e = 0; e < 16; ++e) {
                int c = (e < 8) ? (hi * 8 + e) : (16 + hi * 8 + (e - 8));
                bfr[e] = (_Float16)(q[(size_t)c * PLANE] * okf);
            }
            v8f d = {};
            d = __builtin_amdgcn_wmma_f32_16x16x32_f16(
                false, afr, false, bfr, (short)0, d, false, false);
#pragma unroll
            for (int r = 0; r < 8; ++r) {
                int m = hi * 8 + r;
                sC[m * 48 + t * 16 + px] = d[r];
            }
        }
        __syncthreads();
        // scan this dv row: lane handles pixel m = px, du subset by lane half
        int du0 = (hi == 0) ? -8 : 1;
        int ndu = (hi == 0) ? 9 : 8;
#pragma unroll
        for (int k = 0; k < 9; ++k) {
            if (k < ndu) {
                int du = du0 + k;
                float v = sC[px * 48 + px + du + 16];
                int dIdx = (dv + 8) * 17 + (du + 8);
                if (v > best) { best = v; bestD = dIdx; }
            }
        }
        __syncthreads();
    }

    sBest[L] = best;
    sD[L]    = bestD;
    __syncthreads();

    if (L < 16) {
        float ob = sBest[L + 16];
        int   od = sD[L + 16];
        if (ob > best || (ob == best && od < bestD)) { best = ob; bestD = od; }
        int dvI = bestD / 17, duI = bestD % 17;
        int x = x0 + L;

        float areg[COUT];
#pragma unroll
        for (int c = 0; c < COUT; ++c)
            areg[c] = ap[(size_t)c * PLANE + (size_t)y * HW + x];

        // l2 = 2 - 2*cos on the 3x3 neighborhood (pad 2.0 outside 17x17;
        // out-of-image dots are 0 so the same formula covers both cases)
        float l2n[3][3];
#pragma unroll
        for (int p = 0; p < 3; ++p) {
#pragma unroll
            for (int q2 = 0; q2 < 3; ++q2) {
                int dv2 = dvI + p - 1, du2 = duI + q2 - 1;
                int yy = y + dv2 - 8, xx = x + du2 - 8;
                bool ok = (dv2 >= 0) && (dv2 < 17) && (du2 >= 0) && (du2 < 17) &&
                          (yy >= 0) && (yy < HW) && (xx >= 0) && (xx < HW);
                float mk = ok ? 1.0f : 0.0f;
                int yyc = clampi(yy, 0, HW - 1), xxc = clampi(xx, 0, HW - 1);
                const float* qq = bp + (size_t)yyc * HW + xxc;
                float cv = 0.f;
#pragma unroll
                for (int c = 0; c < COUT; ++c)
                    cv = fmaf(areg[c], qq[(size_t)c * PLANE], cv);
                l2n[p][q2] = 2.0f - 2.0f * cv * mk;
            }
        }

        float hard[4];
#pragma unroll
        for (int k = 0; k < 4; ++k) {
            float s = 0.f;
#pragma unroll
            for (int p = 0; p < 3; ++p)
#pragma unroll
                for (int q2 = 0; q2 < 3; ++q2)
                    s += subAttn[k * 9 + p * 3 + q2] * l2n[p][q2];
            hard[k] = s;
        }
        float subU = fminf(fmaxf(-hard[2] / fmaxf(hard[0], 1e-6f), -1.f), 1.f);
        float subV = fminf(fmaxf(-hard[3] / fmaxf(hard[1], 1e-6f), -1.f), 1.f);

        size_t pixIdx = (size_t)y * HW + x;
        maxValue[(size_t)n * PIX + pixIdx] = best;
        flow[((size_t)n * 2 + 0) * PIX + pixIdx] = (float)(duI - 8) + subU;
        flow[((size_t)n * 2 + 1) * PIX + pixIdx] = (float)(dvI - 8) + subV;
    }
}

// ---------------------------------------------------------------------------
// fold(bwarp(unfold(src,k,pad,st), flow)) / 9 collapsed analytically:
//   out[n,c,y,x] = (1/9) * sum_{ai,aj in 0..2} bilerp over corners of
//     I_{ij}(h',w') = src[c, st*h' + i - st, st*w' + j - st]
//   h = y/st + 1 - ai, i = y%st + ai*st (likewise w, j). All bounds handling
//   is branchless: clamp indices, multiply by mask.
// ---------------------------------------------------------------------------
template <int ST, int C>
__global__ __launch_bounds__(256)
void warp_fold(const float* __restrict__ src, const float* __restrict__ flow,
               float* __restrict__ out) {
    const int R = HW * ST;
    long long id = (long long)blockIdx.x * 256 + threadIdx.x;
    const long long total = 2LL * C * R * R;
    if (id >= total) return;

    int x = (int)(id % R);
    long long t = id / R;
    int y = (int)(t % R); t /= R;
    int c = (int)(t % C);
    int n = (int)(t / C);

    const float* fU = flow + (size_t)n * 2 * PIX;
    const float* fV = fU + PIX;
    const float* sp = src + ((size_t)n * C + c) * (size_t)R * R;

    float acc = 0.f;
    int ybase = y / ST, ymod = y % ST;
    int xbase = x / ST, xmod = x % ST;

#pragma unroll
    for (int ai = 0; ai < 3; ++ai) {
        int h = ybase + 1 - ai;
        int i = ymod + ai * ST;
        float mh = ((h >= 0) && (h < HW)) ? 1.0f : 0.0f;
        int hc = clampi(h, 0, HW - 1);
#pragma unroll
        for (int aj = 0; aj < 3; ++aj) {
            int w = xbase + 1 - aj;
            int j = xmod + aj * ST;
            float mhw = mh * (((w >= 0) && (w < HW)) ? 1.0f : 0.0f);
            int wc = clampi(w, 0, HW - 1);

            float U = fU[hc * HW + wc], V = fV[hc * HW + wc];
            float xf = (float)wc + U, yf = (float)hc + V;
            float x0f = floorf(xf), y0f = floorf(yf);
            float wx = xf - x0f, wy = yf - y0f;
            int x0i = (int)x0f, y0i = (int)y0f;
#pragma unroll
            for (int bb = 0; bb < 4; ++bb) {
                int hp = y0i + (bb >> 1), wp = x0i + (bb & 1);
                float mc = mhw * (((hp >= 0) && (hp < HW) && (wp >= 0) && (wp < HW)) ? 1.0f : 0.0f);
                int hpc = clampi(hp, 0, HW - 1), wpc = clampi(wp, 0, HW - 1);
                int sy = ST * hpc + i - ST, sx = ST * wpc + j - ST;
                float ms = mc * (((sy >= 0) && (sy < R) && (sx >= 0) && (sx < R)) ? 1.0f : 0.0f);
                int syc = clampi(sy, 0, R - 1), sxc = clampi(sx, 0, R - 1);
                float v = sp[(size_t)syc * R + sxc];
                float wgt = ((bb >> 1) ? wy : 1.f - wy) * ((bb & 1) ? wx : 1.f - wx);
                acc = fmaf(v * ms, wgt, acc);
            }
        }
    }
    out[id] = acc * (1.0f / 9.0f);
}

// ---------------------------------------------------------------------------
extern "C" void kernel_launch(void* const* d_in, const int* in_sizes, int n_in,
                              void* d_out, int out_size, void* d_ws, size_t ws_size,
                              hipStream_t stream) {
    (void)in_sizes; (void)n_in; (void)out_size; (void)ws_size;
    // setup_inputs order: Kt1x, Kt2x, KVt4x, V01x, V02x, KV04x, subAttnMatC, W_in
    const float* KVt4x   = (const float*)d_in[2];
    const float* V01x    = (const float*)d_in[3];
    const float* V02x    = (const float*)d_in[4];
    const float* KV04x   = (const float*)d_in[5];
    const float* subAttn = (const float*)d_in[6];
    const float* W_in    = (const float*)d_in[7];

    char* ws = (char*)d_ws;
    _Float16* aPack = (_Float16*)ws;                         //  73,728 B
    float* a_raw = (float*)(ws + 73728);                     // 1,048,576 B
    float* b_raw = (float*)(ws + 73728 + 1048576);           // 1,048,576 B
    float* flow  = (float*)(ws + 73728 + 2 * 1048576);       //    65,536 B

    float* out      = (float*)d_out;
    float* maxValue = out;                                   // 2*1*64*64
    float* T4x      = out + 2 * PIX;                         // 2*128*64*64
    float* T2x      = T4x + 2 * 128 * 64 * 64;               // 2*64*128*128
    float* T1x      = T2x + 2 * 64 * 128 * 128;              // 2*32*256*256

    pack_weights<<<dim3(72), dim3(32), 0, stream>>>(W_in, aPack);
    conv1x1_wmma<<<dim3(1024), dim3(32), 0, stream>>>(KVt4x, KV04x, aPack, a_raw, b_raw);
    l2norm_inplace<<<dim3(64), dim3(256), 0, stream>>>(a_raw, b_raw);
    corr_flow_wmma<<<dim3(512), dim3(32), 0, stream>>>(a_raw, b_raw, subAttn, maxValue, flow);

    warp_fold<1, 128><<<dim3((2 * 128 * 64 * 64) / 256), dim3(256), 0, stream>>>(KV04x, flow, T4x);
    warp_fold<2, 64><<<dim3((2 * 64 * 128 * 128) / 256), dim3(256), 0, stream>>>(V02x, flow, T2x);
    warp_fold<4, 32><<<dim3((2 * 32 * 256 * 256) / 256), dim3(256), 0, stream>>>(V01x, flow, T1x);
}
